// CodeBook_4853313044734
// MI455X (gfx1250) — matrix operand order repaired and checked by reading, not compiled, not run on hardware
//
#include <hip/hip_runtime.h>
#include <hip/hip_bf16.h>

typedef __attribute__((ext_vector_type(16))) _Float16 v16h;
typedef __attribute__((ext_vector_type(8)))  _Float16 v8h;
typedef __attribute__((ext_vector_type(4)))  _Float16 v4h;
typedef __attribute__((ext_vector_type(8)))  float    v8f;
typedef __attribute__((ext_vector_type(4)))  int      v4i;

typedef __attribute__((address_space(3))) void* la_ptr;
typedef __attribute__((address_space(1))) v4i* gv4i_ptr;   // global int4*
typedef __attribute__((address_space(3))) v4i* lv4i_ptr;   // LDS int4*

#define H_DIM 512
#define DIN   1280
#define K_CB  1024
#define BM    128
#define BN    64
#define BK    32

// ---------------------------------------------------------------------------
// CDNA5 async copy global(f16) -> LDS, 16 bytes per lane, ASYNCcnt-tracked.
// ---------------------------------------------------------------------------
#if defined(__has_builtin)
#if __has_builtin(__builtin_amdgcn_global_load_async_to_lds_b128)
#define HAVE_ASYNC_BUILTIN 1
#endif
#if __has_builtin(__builtin_amdgcn_s_wait_asynccnt)
#define HAVE_WAIT_BUILTIN 1
#endif
#endif

static __device__ __forceinline__ void async_copy16(const _Float16* g, _Float16* l)
{
#if defined(HAVE_ASYNC_BUILTIN)
    __builtin_amdgcn_global_load_async_to_lds_b128(
        (gv4i_ptr)(size_t)g, (lv4i_ptr)(la_ptr)l, 0, 0);
#else
    unsigned loff = (unsigned)(size_t)(la_ptr)l;       // LDS byte offset
    asm volatile("global_load_async_to_lds_b128 %0, %1, off"
                 :: "v"(loff), "v"(g) : "memory");
#endif
}

static __device__ __forceinline__ void wait_async0()
{
#if defined(HAVE_WAIT_BUILTIN)
    __builtin_amdgcn_s_wait_asynccnt(0);
#else
    asm volatile("s_wait_asynccnt 0" ::: "memory");
#endif
}

// ---------------------------------------------------------------------------
// WMMA GEMM:  C[M,Nc] = A[M,Kd](f16) * Bt[Nc,Kd]^T(f16)
// epilogue: * rowscale[row], + bias[col], relu; writes f32 C and/or f16 C16.
// Block: 256 thr (8 waves), tile 128x64, K-step 32, double-buffered LDS with
// async-to-LDS prefetch of tile k+1 overlapping the WMMAs of tile k.
// ---------------------------------------------------------------------------
__global__ __launch_bounds__(256)
void gemm_wmma_f16(const _Float16* __restrict__ A, const _Float16* __restrict__ Bt,
                   const float* __restrict__ rowscale, const float* __restrict__ bias,
                   float* __restrict__ C, _Float16* __restrict__ C16,
                   int M, int Kd, int Nc, int relu)
{
    __shared__ _Float16 As[2][BM * BK];
    __shared__ _Float16 Bs[2][BN * BK];

    const int tid  = threadIdx.x;
    const int wave = tid >> 5;
    const int lane = tid & 31;
    const int m0   = blockIdx.x * BM;
    const int n0   = blockIdx.y * BN;
    const int mrow = lane & 15;
    const int hi   = lane >> 4;

    v8f acc[4];
#pragma unroll
    for (int t = 0; t < 4; ++t)
#pragma unroll
        for (int i = 0; i < 8; ++i) acc[t][i] = 0.0f;

    auto stage = [&](int k0, int buf) {
        // A tile: 128x32 halves = 512 x 16B chunks (2 per thread)
#pragma unroll
        for (int it = 0; it < 2; ++it) {
            int chunk = tid + it * 256;
            int r  = chunk >> 2;
            int c8 = (chunk & 3) << 3;
            int gr = m0 + r; gr = gr < M ? gr : (M - 1);   // clamp: padded rows masked at store
            async_copy16(A + (size_t)gr * Kd + k0 + c8, &As[buf][r * BK + c8]);
        }
        // B tile: 64x32 halves = 256 x 16B chunks (1 per thread)
        {
            int n  = tid >> 2;
            int c8 = (tid & 3) << 3;
            async_copy16(Bt + (size_t)(n0 + n) * Kd + k0 + c8, &Bs[buf][n * BK + c8]);
        }
    };

    const int nk = Kd / BK;
    stage(0, 0);
    int cur = 0;
    for (int i = 0; i < nk; ++i) {
        wait_async0();          // current tile's async copies landed in LDS
        __syncthreads();        // visible to all waves; prev buffer free
        if (i + 1 < nk) stage((i + 1) * BK, cur ^ 1);   // prefetch overlaps WMMA

        const _Float16* Ab = As[cur];
        const _Float16* Bb = Bs[cur];
        const int arow = wave * 16 + mrow;
        v8h alo = *(const v8h*)&Ab[arow * BK + hi * 8];        // K 0-7 / 8-15
        v8h ahi = *(const v8h*)&Ab[arow * BK + 16 + hi * 8];   // K 16-23 / 24-31
        v16h af;
#pragma unroll
        for (int j = 0; j < 8; ++j) { af[j] = alo[j]; af[8 + j] = ahi[j]; }

#pragma unroll
        for (int t = 0; t < 4; ++t) {
            const int bcol = t * 16 + mrow;
            v8h blo = *(const v8h*)&Bb[bcol * BK + hi * 8];
            v8h bhi = *(const v8h*)&Bb[bcol * BK + 16 + hi * 8];
            v16h bf;
#pragma unroll
            for (int j = 0; j < 8; ++j) { bf[j] = blo[j]; bf[8 + j] = bhi[j]; }
            acc[t] = __builtin_amdgcn_wmma_f32_16x16x32_f16(
                false, af, false, bf, (short)0, acc[t], false, false);
        }
        cur ^= 1;
    }

#pragma unroll
    for (int t = 0; t < 4; ++t) {
        int col = n0 + t * 16 + mrow;
#pragma unroll
        for (int r = 0; r < 8; ++r) {
            int row = m0 + wave * 16 + r + hi * 8;
            if (row < M) {
                float v = acc[t][r];
                if (rowscale) v *= rowscale[row];
                if (bias) v += bias[col];
                if (relu) v = fmaxf(v, 0.f);
                if (C)   C[(size_t)row * Nc + col] = v;
                if (C16) C16[(size_t)row * Nc + col] = (_Float16)v;
            }
        }
    }
}

// ---------------------------------------------------------------------------
// Elementwise / graph / reduction kernels
// ---------------------------------------------------------------------------
__global__ void fill_zero(float* __restrict__ p, long n) {
    for (long i = blockIdx.x * (long)blockDim.x + threadIdx.x; i < n;
         i += (long)gridDim.x * blockDim.x) p[i] = 0.f;
}

__global__ void copy_f32(const float* __restrict__ s, float* __restrict__ d, long n) {
    for (long i = blockIdx.x * (long)blockDim.x + threadIdx.x; i < n;
         i += (long)gridDim.x * blockDim.x) d[i] = s[i];
}

__global__ void f32_to_f16(const float* __restrict__ s, _Float16* __restrict__ d, long n) {
    for (long i = blockIdx.x * (long)blockDim.x + threadIdx.x; i < n;
         i += (long)gridDim.x * blockDim.x) d[i] = (_Float16)s[i];
}

// W[R,C] f32 -> Wt[C,R] f16 (pre-transposed weights for the GEMM's B path)
__global__ void transp_f16(const float* __restrict__ W, _Float16* __restrict__ Wt,
                           int R, int Cc) {
    long total = (long)R * Cc;
    for (long i = blockIdx.x * (long)blockDim.x + threadIdx.x; i < total;
         i += (long)gridDim.x * blockDim.x) {
        int r = (int)(i / Cc), c = (int)(i % Cc);
        Wt[(size_t)c * R + r] = (_Float16)W[i];
    }
}

__global__ void count_deg(const int* __restrict__ idx, float* __restrict__ deg, int E) {
    int i = blockIdx.x * 256 + threadIdx.x;
    if (i < E) unsafeAtomicAdd(&deg[idx[i]], 1.0f);
}

__global__ void finalize_norm(float* __restrict__ deg, long n) {
    for (long i = blockIdx.x * (long)blockDim.x + threadIdx.x; i < n;
         i += (long)gridDim.x * blockDim.x)
        deg[i] = rsqrtf(fmaxf(deg[i], 1.0f));
}

__global__ void init_bias3(float* __restrict__ out, const float* __restrict__ b0,
                           const float* __restrict__ b1, const float* __restrict__ b2,
                           long total) {
    for (long i = blockIdx.x * (long)blockDim.x + threadIdx.x; i < total;
         i += (long)gridDim.x * blockDim.x) {
        int c = (int)(i % H_DIM);
        out[i] = b0[c] + b1[c] + b2[c];
    }
}

// one edge per block; out[dst] += P16[src] * inn[dst]   (f16 read, f32 atomics)
__global__ __launch_bounds__(128)
void scatter_edges(const _Float16* __restrict__ P16, const int* __restrict__ src,
                   const int* __restrict__ dst, const float* __restrict__ inn,
                   float* __restrict__ out, int E)
{
    int e = blockIdx.x;
    if (e >= E) return;
    int s = src[e], d = dst[e];
    float w = inn[d];
    const v4h* ps = (const v4h*)(P16 + (size_t)s * H_DIM);
    float* po = out + (size_t)d * H_DIM;
    __builtin_prefetch(ps, 0, 1);
    int c = threadIdx.x;                       // 0..127, 4 channels each
    v4h v = ps[c];
    unsafeAtomicAdd(po + 4 * c + 0, (float)v[0] * w);
    unsafeAtomicAdd(po + 4 * c + 1, (float)v[1] * w);
    unsafeAtomicAdd(po + 4 * c + 2, (float)v[2] * w);
    unsafeAtomicAdd(po + 4 * c + 3, (float)v[3] * w);
}

__global__ __launch_bounds__(256)
void bn_stats(const float* __restrict__ h, float* __restrict__ stats, int M) {
    int c = blockIdx.x;
    float s1 = 0.f, s2 = 0.f;
    for (int r = threadIdx.x; r < M; r += 256) {
        float v = h[(size_t)r * H_DIM + c];
        s1 += v; s2 += v * v;
    }
    __shared__ float r1[256], r2[256];
    r1[threadIdx.x] = s1; r2[threadIdx.x] = s2;
    __syncthreads();
    for (int s = 128; s > 0; s >>= 1) {
        if (threadIdx.x < s) { r1[threadIdx.x] += r1[threadIdx.x + s];
                               r2[threadIdx.x] += r2[threadIdx.x + s]; }
        __syncthreads();
    }
    if (threadIdx.x == 0) {
        float mu  = r1[0] / (float)M;
        float var = r2[0] / (float)M - mu * mu;
        stats[c]         = mu;
        stats[H_DIM + c] = rsqrtf(var + 1e-5f);
    }
}

// BN normalize; keeps f32 in place and emits f16 copy for the next WMMA stage
__global__ void bn_apply(float* __restrict__ h, const float* __restrict__ stats,
                         const float* __restrict__ g, const float* __restrict__ b,
                         _Float16* __restrict__ out16, long total) {
    for (long i = blockIdx.x * (long)blockDim.x + threadIdx.x; i < total;
         i += (long)gridDim.x * blockDim.x) {
        int c = (int)(i % H_DIM);
        float v = (h[i] - stats[c]) * stats[H_DIM + c] * g[c] + b[c];
        h[i] = v;
        out16[i] = (_Float16)v;
    }
}

__global__ __launch_bounds__(256)
void l2norm_rows(const float* __restrict__ src, float* __restrict__ dst,
                 _Float16* __restrict__ dst16, int R, int Cn) {
    int row = blockIdx.x * 8 + (threadIdx.x >> 5);
    int lane = threadIdx.x & 31;
    if (row >= R) return;
    const float* s = src + (size_t)row * Cn;
    float acc = 0.f;
    for (int c = lane; c < Cn; c += 32) { float v = s[c]; acc += v * v; }
    for (int off = 16; off > 0; off >>= 1) acc += __shfl_xor(acc, off, 32);
    float inv = 1.0f / fmaxf(sqrtf(acc), 1e-12f);
    for (int c = lane; c < Cn; c += 32) {
        float v = s[c] * inv;
        dst[(size_t)row * Cn + c] = v;
        if (dst16) dst16[(size_t)row * Cn + c] = (_Float16)v;
    }
}

__global__ __launch_bounds__(256)
void argmax_rows(const float* __restrict__ S, int* __restrict__ idx, int R, int Cn) {
    int row = blockIdx.x * 8 + (threadIdx.x >> 5);
    int lane = threadIdx.x & 31;
    if (row >= R) return;
    const float* s = S + (size_t)row * Cn;
    float best = -3.0e38f; int bi = 0;
    for (int c = lane; c < Cn; c += 32) {
        float v = s[c];
        if (v > best) { best = v; bi = c; }
    }
    for (int off = 16; off > 0; off >>= 1) {
        float ov = __shfl_xor(best, off, 32);
        int   oi = __shfl_xor(bi, off, 32);
        if (ov > best || (ov == best && oi < bi)) { best = ov; bi = oi; }
    }
    if (lane == 0) idx[row] = bi;
}

__global__ void zero_small(float* __restrict__ a, float* __restrict__ b) {
    if (threadIdx.x < 3) a[threadIdx.x] = 0.f;
    if (threadIdx.x < 2) b[threadIdx.x] = 0.f;
}

// q = cbn[idx]; e16 = q; em16/emaskf = masked q; accumulate 1.25*mean((q-zn)^2)
__global__ __launch_bounds__(256)
void vq_gather(const float* __restrict__ cbn, const float* __restrict__ zn,
               const int* __restrict__ idx, const unsigned char* __restrict__ mask,
               _Float16* __restrict__ e16, _Float16* __restrict__ em16,
               float* __restrict__ emaskf, float* __restrict__ eq_accum,
               int Nn, float scale)
{
    int n = blockIdx.x;
    if (n >= Nn) return;
    int code = idx[n];
    int mflag = mask[code] != 0;
    const float* q = cbn + (size_t)code * H_DIM;
    const float* z = zn + (size_t)n * H_DIM;
    float part = 0.f;
    for (int c = threadIdx.x; c < H_DIM; c += 256) {
        float qv = q[c];
        float d = qv - z[c];
        part += d * d;
        float mv = mflag ? 0.f : qv;
        e16[(size_t)n * H_DIM + c]    = (_Float16)qv;
        em16[(size_t)n * H_DIM + c]   = (_Float16)mv;
        emaskf[(size_t)n * H_DIM + c] = mv;
    }
    __shared__ float red[256];
    red[threadIdx.x] = part;
    __syncthreads();
    for (int s = 128; s > 0; s >>= 1) {
        if (threadIdx.x < s) red[threadIdx.x] += red[threadIdx.x + s];
        __syncthreads();
    }
    if (threadIdx.x == 0) unsafeAtomicAdd(eq_accum, red[0] * scale);
}

__global__ __launch_bounds__(256)
void mse_reduce(const float* __restrict__ a, const float* __restrict__ b,
                float* __restrict__ accum, long total, float scale) {
    float part = 0.f;
    for (long i = blockIdx.x * (long)blockDim.x + threadIdx.x; i < total;
         i += (long)gridDim.x * blockDim.x) {
        float d = a[i] - b[i];
        part += d * d;
    }
    __shared__ float red[256];
    red[threadIdx.x] = part;
    __syncthreads();
    for (int s = 128; s > 0; s >>= 1) {
        if (threadIdx.x < s) red[threadIdx.x] += red[threadIdx.x + s];
        __syncthreads();
    }
    if (threadIdx.x == 0) unsafeAtomicAdd(accum, red[0] * scale);
}

__global__ __launch_bounds__(256)
void mask_loss_kernel(const float* __restrict__ xrec, const float* __restrict__ x,
                      const int* __restrict__ idx, const unsigned char* __restrict__ mask,
                      float* __restrict__ num, float* __restrict__ den, int Nn)
{
    int row = blockIdx.x * 8 + (threadIdx.x >> 5);
    int lane = threadIdx.x & 31;
    if (row >= Nn) return;
    if (mask[idx[row]] == 0) return;
    const float* a = xrec + (size_t)row * DIN;
    const float* b = x + (size_t)row * DIN;
    float dot = 0.f, na = 0.f, nb = 0.f;
    for (int c = lane; c < DIN; c += 32) {
        float av = a[c], bv = b[c];
        dot += av * bv; na += av * av; nb += bv * bv;
    }
    for (int off = 16; off > 0; off >>= 1) {
        dot += __shfl_xor(dot, off, 32);
        na  += __shfl_xor(na, off, 32);
        nb  += __shfl_xor(nb, off, 32);
    }
    if (lane == 0) {
        float inv = 1.0f / (fmaxf(sqrtf(na), 1e-12f) * fmaxf(sqrtf(nb), 1e-12f));
        float per = 1.0f - dot * inv;
        unsafeAtomicAdd(num, per * per);
        unsafeAtomicAdd(den, 1.0f);
    }
}

__global__ void finalize_mask(const float* __restrict__ scal, float* __restrict__ out) {
    out[0] = scal[0] / (scal[1] + 1e-12f);
}

// ---------------------------------------------------------------------------
// Host orchestration
// ---------------------------------------------------------------------------
extern "C" void kernel_launch(void* const* d_in, const int* in_sizes, int n_in,
                              void* d_out, int out_size, void* d_ws, size_t ws_size,
                              hipStream_t stream)
{
    (void)n_in; (void)out_size; (void)ws_size;
    const float* x = (const float*)d_in[0];
    const int* srcp[3] = {(const int*)d_in[1], (const int*)d_in[3], (const int*)d_in[5]};
    const int* dstp[3] = {(const int*)d_in[2], (const int*)d_in[4], (const int*)d_in[6]};
    const float* enc_conv_w0 = (const float*)d_in[7];
    const float* enc_conv_b0 = (const float*)d_in[8];
    const float* enc_conv_w  = (const float*)d_in[9];
    const float* enc_conv_b  = (const float*)d_in[10];
    const float* enc_fc_w    = (const float*)d_in[11];
    const float* enc_fc_b    = (const float*)d_in[12];
    const float* enc_bn_g    = (const float*)d_in[13];
    const float* enc_bn_b    = (const float*)d_in[14];
    const float* dec_conv_w  = (const float*)d_in[15];
    const float* dec_conv_b  = (const float*)d_in[16];
    const float* dec_fc_w    = (const float*)d_in[17];
    const float* dec_fc_b    = (const float*)d_in[18];
    const float* dec_fc_lw   = (const float*)d_in[19];
    const float* dec_fc_lb   = (const float*)d_in[20];
    const float* dec_bn_g    = (const float*)d_in[21];
    const float* dec_bn_b    = (const float*)d_in[22];
    const float* codebook    = (const float*)d_in[23];
    const unsigned char* maskp = (const unsigned char*)d_in[24];

    const int N = in_sizes[0] / DIN;
    const int E = in_sizes[1];
    const long NH = (long)N * H_DIM;

    // ---- workspace layout
    char* wcur = (char*)d_ws;
    auto alloc = [&](size_t bytes) -> void* {
        void* p = (void*)wcur;
        wcur += (bytes + 255) & ~(size_t)255;
        return p;
    };
    float* norms = (float*)alloc(6 * (size_t)N * 4);       // on0,inn0,on1,inn1,on2,inn2
    float* hA    = (float*)alloc(NH * 4);
    float* hB    = (float*)alloc(NH * 4);
    float* S     = (float*)alloc((size_t)N * K_CB * 4);
    float* xrec  = (float*)alloc((size_t)N * DIN * 4);
    float* zn    = (float*)alloc(NH * 4);
    float* cbn   = (float*)alloc((size_t)K_CB * H_DIM * 4);
    int*   idx   = (int*)alloc((size_t)N * 4);
    float* stats = (float*)alloc(2 * H_DIM * 4);
    float* scal  = (float*)alloc(64);                      // [num, den]
    _Float16* x16  = (_Float16*)alloc((size_t)N * DIN * 2);
    _Float16* h16  = (_Float16*)alloc(NH * 2);
    _Float16* P16  = (_Float16*)alloc(NH * 2);
    _Float16* e16  = (_Float16*)alloc(NH * 2);
    _Float16* em16 = (_Float16*)alloc(NH * 2);
    _Float16* zn16 = (_Float16*)alloc(NH * 2);
    _Float16* cbn16 = (_Float16*)alloc((size_t)K_CB * H_DIM * 2);
    _Float16* wt_conv_enc[3][3];
    _Float16* wt_fc_enc[3];
    _Float16* wt_conv_dec[3][3];
    _Float16* wt_fc_dec[2];
    _Float16* wt_last;
    for (int r = 0; r < 3; ++r) wt_conv_enc[0][r] = (_Float16*)alloc((size_t)DIN * H_DIM * 2);
    for (int l = 1; l < 3; ++l)
        for (int r = 0; r < 3; ++r) wt_conv_enc[l][r] = (_Float16*)alloc((size_t)H_DIM * H_DIM * 2);
    for (int l = 0; l < 3; ++l) wt_fc_enc[l] = (_Float16*)alloc((size_t)H_DIM * H_DIM * 2);
    for (int l = 0; l < 3; ++l)
        for (int r = 0; r < 3; ++r) wt_conv_dec[l][r] = (_Float16*)alloc((size_t)H_DIM * H_DIM * 2);
    for (int l = 0; l < 2; ++l) wt_fc_dec[l] = (_Float16*)alloc((size_t)H_DIM * H_DIM * 2);
    wt_last = (_Float16*)alloc((size_t)H_DIM * DIN * 2);

    float* outf  = (float*)d_out;
    float* z_out = outf;
    float* emask = outf + NH;
    float* sc    = outf + 2 * NH;                          // [e_q, recon, mask]

    const dim3 gH((N + BM - 1) / BM, H_DIM / BN);
    const dim3 gK((N + BM - 1) / BM, K_CB / BN);
    const dim3 gD((N + BM - 1) / BM, DIN / BN);
    const int eb = (E + 255) / 256;

    // ---- one-time per launch: f16 weight arena (pre-transposed to [Nc,Kd])
    auto cvtT = [&](const float* W, _Float16* Wt, int R, int Cc) {
        transp_f16<<<1024, 256, 0, stream>>>(W, Wt, R, Cc);
    };
    for (int r = 0; r < 3; ++r)
        cvtT(enc_conv_w0 + (size_t)r * DIN * H_DIM, wt_conv_enc[0][r], DIN, H_DIM);
    for (int l = 1; l < 3; ++l)
        for (int r = 0; r < 3; ++r)
            cvtT(enc_conv_w + (size_t)((l - 1) * 3 + r) * H_DIM * H_DIM,
                 wt_conv_enc[l][r], H_DIM, H_DIM);
    for (int l = 0; l < 3; ++l)
        cvtT(enc_fc_w + (size_t)l * H_DIM * H_DIM, wt_fc_enc[l], H_DIM, H_DIM);
    for (int l = 0; l < 3; ++l)
        for (int r = 0; r < 3; ++r)
            cvtT(dec_conv_w + (size_t)(l * 3 + r) * H_DIM * H_DIM,
                 wt_conv_dec[l][r], H_DIM, H_DIM);
    for (int l = 0; l < 2; ++l)
        cvtT(dec_fc_w + (size_t)l * H_DIM * H_DIM, wt_fc_dec[l], H_DIM, H_DIM);
    cvtT(dec_fc_lw, wt_last, H_DIM, DIN);
    f32_to_f16<<<4096, 256, 0, stream>>>(x, x16, (long)N * DIN);

    // ---- degree norms (deg^{-1/2}, clamp >= 1)
    fill_zero<<<2048, 256, 0, stream>>>(norms, 6L * N);
    for (int r = 0; r < 3; ++r) {
        count_deg<<<eb, 256, 0, stream>>>(srcp[r], norms + (size_t)(2 * r) * N, E);
        count_deg<<<eb, 256, 0, stream>>>(dstp[r], norms + (size_t)(2 * r + 1) * N, E);
    }
    finalize_norm<<<2048, 256, 0, stream>>>(norms, 6L * N);

    // ---- hetero conv: out = sum_r scatter(on_r ⊙ (h @ W_r), dst_r)*inn_r + Σb_r
    auto hetero = [&](const _Float16* a16, int Kin, _Float16* const* WtR,
                      const float* Bb, float* outp) {
        init_bias3<<<4096, 256, 0, stream>>>(outp, Bb, Bb + H_DIM, Bb + 2 * H_DIM, NH);
        for (int r = 0; r < 3; ++r) {
            gemm_wmma_f16<<<gH, 256, 0, stream>>>(a16, WtR[r],
                                                  norms + (size_t)(2 * r) * N, nullptr,
                                                  nullptr, P16, N, Kin, H_DIM, 0);
            scatter_edges<<<E, 128, 0, stream>>>(P16, srcp[r], dstp[r],
                                                 norms + (size_t)(2 * r + 1) * N,
                                                 outp, E);
        }
    };

    // ================= encoder =================
    for (int l = 0; l < 3; ++l) {
        const _Float16* a16 = (l == 0) ? x16 : h16;
        int Kin = (l == 0) ? DIN : H_DIM;
        const float* Bb = (l == 0) ? enc_conv_b0 : enc_conv_b + (size_t)(l - 1) * 3 * H_DIM;
        hetero(a16, Kin, wt_conv_enc[l], Bb, hB);
        f32_to_f16<<<4096, 256, 0, stream>>>(hB, h16, NH);
        gemm_wmma_f16<<<gH, 256, 0, stream>>>(h16, wt_fc_enc[l], nullptr,
                                              enc_fc_b + (size_t)l * H_DIM,
                                              hA, nullptr, N, H_DIM, H_DIM, 1);
        bn_stats<<<H_DIM, 256, 0, stream>>>(hA, stats, N);
        bn_apply<<<4096, 256, 0, stream>>>(hA, stats, enc_bn_g + (size_t)l * H_DIM,
                                           enc_bn_b + (size_t)l * H_DIM, h16, NH);
    }
    copy_f32<<<4096, 256, 0, stream>>>(hA, z_out, NH);     // z

    // ================= VQ =================
    l2norm_rows<<<(N + 7) / 8, 256, 0, stream>>>(hA, zn, zn16, N, H_DIM);
    l2norm_rows<<<(K_CB + 7) / 8, 256, 0, stream>>>(codebook, cbn, cbn16, K_CB, H_DIM);
    gemm_wmma_f16<<<gK, 256, 0, stream>>>(zn16, cbn16, nullptr, nullptr,
                                          S, nullptr, N, H_DIM, K_CB, 0);
    argmax_rows<<<(N + 7) / 8, 256, 0, stream>>>(S, idx, N, K_CB);
    zero_small<<<1, 64, 0, stream>>>(sc, scal);
    vq_gather<<<N, 256, 0, stream>>>(cbn, zn, idx, maskp, e16, em16, emask,
                                     sc + 0, N, 1.25f / (float)NH);

    // ================= decoder (x2) =================
    auto decode = [&](const _Float16* hin16) {
        const _Float16* a16 = hin16;
        for (int l = 0; l < 3; ++l) {
            hetero(a16, H_DIM, wt_conv_dec[l], dec_conv_b + (size_t)l * 3 * H_DIM, hB);
            f32_to_f16<<<4096, 256, 0, stream>>>(hB, h16, NH);
            if (l != 2) {
                gemm_wmma_f16<<<gH, 256, 0, stream>>>(h16, wt_fc_dec[l], nullptr,
                                                      dec_fc_b + (size_t)l * H_DIM,
                                                      hA, nullptr, N, H_DIM, H_DIM, 1);
                bn_stats<<<H_DIM, 256, 0, stream>>>(hA, stats, N);
                bn_apply<<<4096, 256, 0, stream>>>(hA, stats,
                                                   dec_bn_g + (size_t)l * H_DIM,
                                                   dec_bn_b + (size_t)l * H_DIM, h16, NH);
                a16 = h16;
            } else {
                gemm_wmma_f16<<<gD, 256, 0, stream>>>(h16, wt_last, nullptr, dec_fc_lb,
                                                      xrec, nullptr, N, H_DIM, DIN, 0);
            }
        }
    };

    decode(e16);
    mse_reduce<<<2048, 256, 0, stream>>>(xrec, x, sc + 1, (long)N * DIN,
                                         1.0f / (float)((long)N * DIN));
    decode(em16);
    mask_loss_kernel<<<(N + 7) / 8, 256, 0, stream>>>(xrec, x, idx, maskp,
                                                      scal + 0, scal + 1, N);
    finalize_mask<<<1, 1, 0, stream>>>(scal, sc + 2);
}
